// MochiAttention_67199058313255
// MI455X (gfx1250) — compile-verified, matrix operand order checked by reading, not run on hardware
//
#include <hip/hip_runtime.h>
#include <cstdint>
#include <cstddef>

// ---------------------------------------------------------------------------
// Problem constants (B=1)
// ---------------------------------------------------------------------------
#define H_HEADS 24
#define EDIM    128
#define DMODEL  3072      // H*E
#define NSPA    2048      // spatial tokens
#define LPRM    256       // prompt tokens
#define PDIM    1536      // added_kv_proj_dim
#define STOT    2304      // NSPA + LPRM
#define QK3     9216      // 3*DMODEL

typedef __attribute__((ext_vector_type(16))) __bf16 v16bf;
typedef __attribute__((ext_vector_type(8)))  __bf16 v8bf;
typedef __attribute__((ext_vector_type(8)))  float  v8f;
typedef __attribute__((ext_vector_type(4)))  unsigned int u32x4;
typedef __attribute__((ext_vector_type(8)))  unsigned int u32x8;

__device__ __forceinline__ unsigned short f2bf(float f) {
  union { float f; unsigned int u; } x; x.f = f;
  unsigned int u = x.u;
  unsigned int r = u + 0x7FFFu + ((u >> 16) & 1u);   // round-to-nearest-even
  return (unsigned short)(r >> 16);
}

// 16x32 bf16 WMMA operand fragment (A-layout; B uses the mirrored layout on a
// transposed source so the same loader serves both):
//   lane L -> row (L & 15); kb = (L & 16) ? 8 : 0
//   elements 0..7  = [kb, kb+8) ; elements 8..15 = [kb+16, kb+24)
__device__ __forceinline__ v16bf load_frag(const unsigned short* base, int ld, int lane) {
  int r  = lane & 15;
  int kb = (lane & 16) ? 8 : 0;
  const unsigned short* p = base + (size_t)r * ld + kb;
  union { v16bf v; v8bf h[2]; } u;
  u.h[0] = *reinterpret_cast<const v8bf*>(p);
  u.h[1] = *reinterpret_cast<const v8bf*>(p + 16);
  return u.v;
}

__device__ __forceinline__ v8f wmma_bf16(v16bf a, v16bf b, v8f c) {
  return __builtin_amdgcn_wmma_f32_16x16x32_bf16(false, a, false, b, (short)0, c, false, false);
}

__device__ __forceinline__ v8f v8f_zero() {
  v8f z = {0.f, 0.f, 0.f, 0.f, 0.f, 0.f, 0.f, 0.f};
  return z;
}

// ---------------------------------------------------------------------------
// CDNA5 async data movement helpers
// ---------------------------------------------------------------------------
// Generic pointers to LDS are {SHARED_BASE[63:32], lds_offset[31:0]} -> the low
// 32 bits are the LDS byte address used by async/TDM instructions.
__device__ __forceinline__ unsigned lds_off_of(const void* p) {
  return (unsigned)(uintptr_t)p;
}

// Per-lane async global->LDS 16B copy (ASYNCcnt tracked).
__device__ __forceinline__ void async_load_b128(unsigned lds_off, const void* gptr) {
  unsigned long long ga = (unsigned long long)(uintptr_t)gptr;
  asm volatile("global_load_async_to_lds_b128 %0, %1, off"
               :: "v"(lds_off), "v"(ga)
               : "memory");
}

__device__ __forceinline__ void wait_async0() {
  asm volatile("s_wait_asynccnt 0x0" ::: "memory");
}
__device__ __forceinline__ void wait_tensor0() {
  asm volatile("s_wait_tensorcnt 0x0" ::: "memory");
}

// Tensor Data Mover: 2D tile load (bf16 elements, data_size=2B), 2-group D#.
// Loads tile_d1 rows of tile_d0 elements, row stride d0_stride (elements),
// optional LDS padding: after every (2<<pad_i) DWORDs insert (pad_a+1) DWORDs.
__device__ __forceinline__ void tdm_load_2d_bf16(unsigned lds_off, const void* gptr,
                                                 unsigned tensor_d0, unsigned tensor_d1,
                                                 unsigned d0_stride,
                                                 unsigned tile_d0, unsigned tile_d1,
                                                 unsigned pad_en, unsigned pad_i,
                                                 unsigned pad_a) {
  unsigned long long ga = (unsigned long long)(uintptr_t)gptr;
  u32x4 g0 = {
      1u,                                                   // count=1 (valid D#)
      lds_off,                                              // lds_addr
      (unsigned)(ga & 0xFFFFFFFFu),                         // global_addr[31:0]
      (unsigned)((ga >> 32) & 0x01FFFFFFu) | (2u << 30)     // ga[56:32] | type=2
  };
  u32x8 g1 = {
      (1u << 16) | (pad_en << 20) | (pad_i << 22) | (pad_a << 25), // data_size=2B
      (tensor_d0 & 0xFFFFu) << 16,                          // tensor_dim0[15:0]
      (tensor_d0 >> 16) | ((tensor_d1 & 0xFFFFu) << 16),    // td0[31:16], td1[15:0]
      (tensor_d1 >> 16) | (tile_d0 << 16),                  // td1[31:16], tile_dim0
      (tile_d1 & 0xFFFFu),                                  // tile_dim1 (tile_dim2=0)
      d0_stride,                                            // tensor_dim0_stride[31:0]
      0u,                                                   // stride[47:32], d1_stride lo
      0u
  };
  asm volatile("tensor_load_to_lds %0, %1" :: "s"(g0), "s"(g1) : "memory");
}

// ---------------------------------------------------------------------------
// Elementwise fp32 -> bf16 cast
// ---------------------------------------------------------------------------
__global__ void cast_f32_bf16(const float* __restrict__ x, unsigned short* __restrict__ y,
                              size_t n) {
  size_t i = (size_t)blockIdx.x * 256 + threadIdx.x;
  if (i < n) y[i] = f2bf(x[i]);
}

// ---------------------------------------------------------------------------
// Transpose + cast: W[K,N] fp32 -> Wt[N,K] bf16 (LDS-tiled, 32x32)
// ---------------------------------------------------------------------------
__global__ __launch_bounds__(256) void transpose_cast(const float* __restrict__ W,
                                                      unsigned short* __restrict__ Wt,
                                                      int K, int Nn) {
  __shared__ float tile[32][33];
  int n0 = blockIdx.x * 32;
  int k0 = blockIdx.y * 32;
  int tx = threadIdx.x;        // 0..31
  int ty = threadIdx.y;        // 0..7
#pragma unroll
  for (int i = 0; i < 4; ++i) {
    int k = k0 + ty + i * 8;
    tile[ty + i * 8][tx] = W[(size_t)k * Nn + n0 + tx];
  }
  __syncthreads();
#pragma unroll
  for (int i = 0; i < 4; ++i) {
    int n = n0 + ty + i * 8;
    Wt[(size_t)n * K + k0 + tx] = f2bf(tile[tx][ty + i * 8]);
  }
}

// ---------------------------------------------------------------------------
// bf16 GEMM with fp32 accumulation:
//   C[M,Nn] = A[M,K] @ Bt[Nn,K]^T (+ bias[Nn])
// Block: 256 threads (8 waves), tile BM=128 x BN=128, BK=32, double-buffered.
// A tile staged with per-lane GLOBAL_LOAD_ASYNC_TO_LDS_B128; B (weight) tile
// staged by the Tensor Data Mover (wave 0 issues a 2D descriptor whose LDS
// pad settings reproduce the padded stride LDT=40 bf16).
// ---------------------------------------------------------------------------
#define BM 128
#define BN 128
#define BK 32
#define LDT 40   // padded LDS row stride (bf16): 64B data + 16B pad per row

__global__ __launch_bounds__(256) void gemm_bf16(const unsigned short* __restrict__ A,
                                                 const unsigned short* __restrict__ Bt,
                                                 const float* __restrict__ bias,
                                                 float* __restrict__ C,
                                                 int M, int Nn, int K, int has_bias) {
  __shared__ unsigned short As[2 * BM * LDT];
  __shared__ unsigned short Bs[2 * BN * LDT];

  int tid  = threadIdx.x;
  int lane = tid & 31;
  int wave = tid >> 5;
  int wm   = wave & 3;   // 32-row strip
  int wn   = wave >> 2;  // 64-col strip
  int bm   = blockIdx.y * BM;
  int bn   = blockIdx.x * BN;

  unsigned asb = lds_off_of(As);
  unsigned bsb = lds_off_of(Bs);

  // this thread's two 16B staging chunks of the A tile
  int c0   = tid * 2;
  int ar0  = c0 >> 2;
  int ac0  = (c0 & 3) * 8;
  int ar1  = (c0 + 1) >> 2;
  int ac1  = ((c0 + 1) & 3) * 8;

  v8f acc[2][4];
#pragma unroll
  for (int mt = 0; mt < 2; ++mt)
#pragma unroll
    for (int nt = 0; nt < 4; ++nt) acc[mt][nt] = v8f_zero();

  const int nk = K / BK;

  // ---- stage tile kt into buffer buf ----
  auto stage = [&](int kt, int buf) {
    int k0 = kt * BK;
    unsigned abase = asb + (unsigned)buf * (BM * LDT * 2);
    async_load_b128(abase + ar0 * (LDT * 2) + ac0 * 2,
                    A + (size_t)(bm + ar0) * K + k0 + ac0);
    async_load_b128(abase + ar1 * (LDT * 2) + ac1 * 2,
                    A + (size_t)(bm + ar1) * K + k0 + ac1);
    if (wave == 0) {
      // TDM: 128 rows x 32 bf16, row stride K; pad 4 DWORDs after every
      // 16 DWORDs -> LDS row stride = 40 bf16 = LDT.
      tdm_load_2d_bf16(bsb + (unsigned)buf * (BN * LDT * 2),
                       Bt + (size_t)bn * K + k0,
                       (unsigned)K, BN, (unsigned)K,
                       BK, BN,
                       1u, 3u, 3u);
    }
  };

  stage(0, 0);
  wait_async0();
  wait_tensor0();
  __syncthreads();

  for (int kt = 0; kt < nk; ++kt) {
    int cur = kt & 1;
    if (kt + 1 < nk) stage(kt + 1, cur ^ 1);

    const unsigned short* Ab = As + cur * (BM * LDT);
    const unsigned short* Bb = Bs + cur * (BN * LDT);
    v16bf aop[2], bop[4];
#pragma unroll
    for (int mt = 0; mt < 2; ++mt)
      aop[mt] = load_frag(Ab + (wm * 32 + mt * 16) * LDT, LDT, lane);
#pragma unroll
    for (int nt = 0; nt < 4; ++nt)
      bop[nt] = load_frag(Bb + (wn * 64 + nt * 16) * LDT, LDT, lane);

#pragma unroll
    for (int mt = 0; mt < 2; ++mt)
#pragma unroll
      for (int nt = 0; nt < 4; ++nt)
        acc[mt][nt] = wmma_bf16(aop[mt], bop[nt], acc[mt][nt]);

    wait_async0();
    wait_tensor0();
    __syncthreads();
  }

  // Epilogue: C layout -> row = r + 8*(lane>=16), col = lane&15 per 16x16 tile
  int half = lane >> 4;
  int lcol = lane & 15;
#pragma unroll
  for (int mt = 0; mt < 2; ++mt)
#pragma unroll
    for (int nt = 0; nt < 4; ++nt) {
      int colg = bn + wn * 64 + nt * 16 + lcol;
      float bv = has_bias ? bias[colg] : 0.f;
#pragma unroll
      for (int r = 0; r < 8; ++r) {
        int rowg = bm + wm * 32 + mt * 16 + r + half * 8;
        C[(size_t)rowg * Nn + colg] = acc[mt][nt][r] + bv;
      }
    }
}

// ---------------------------------------------------------------------------
// QKV epilogue: per (token s, head h) block of 128 threads.
// RMSNorm(q,k) per head, RoPE (spatial only), fold 1/sqrt(E) into q.
// Writes Qf/Kf as bf16 [H,STOT,E]; V transposed as bf16 [H,E,STOT].
// ---------------------------------------------------------------------------
__global__ __launch_bounds__(128) void qkv_epilogue(
    const float* __restrict__ qkv,   // [NSPA, 9216]
    const float* __restrict__ aqkv,  // [LPRM, 9216]
    const float* __restrict__ nqw, const float* __restrict__ nkw,
    const float* __restrict__ naqw, const float* __restrict__ nakw,
    const float* __restrict__ rc, const float* __restrict__ rs,  // [NSPA,128]
    unsigned short* __restrict__ Qf, unsigned short* __restrict__ Kf,
    unsigned short* __restrict__ Vt) {
  int e = threadIdx.x;     // 0..127
  int s = blockIdx.x;      // 0..STOT-1
  int h = blockIdx.y;      // 0..23
  bool spatial = (s < NSPA);
  const float* row = spatial ? (qkv + (size_t)s * QK3)
                             : (aqkv + (size_t)(s - NSPA) * QK3);
  float qv = row[h * EDIM + e];
  float kv = row[DMODEL + h * EDIM + e];
  float vv = row[2 * DMODEL + h * EDIM + e];

  __shared__ float red[128];
  __shared__ float qsh[128];
  __shared__ float ksh[128];

  red[e] = qv * qv;
  __syncthreads();
#pragma unroll
  for (int st = 64; st > 0; st >>= 1) {
    if (e < st) red[e] += red[e + st];
    __syncthreads();
  }
  float qn = qv * rsqrtf(red[0] * (1.0f / 128.0f) + 1e-5f) *
             (spatial ? nqw[e] : naqw[e]);
  __syncthreads();

  red[e] = kv * kv;
  __syncthreads();
#pragma unroll
  for (int st = 64; st > 0; st >>= 1) {
    if (e < st) red[e] += red[e + st];
    __syncthreads();
  }
  float kn = kv * rsqrtf(red[0] * (1.0f / 128.0f) + 1e-5f) *
             (spatial ? nkw[e] : nakw[e]);
  __syncthreads();

  qsh[e] = qn;
  ksh[e] = kn;
  __syncthreads();
  if (spatial) {
    float c  = rc[(size_t)s * EDIM + e];
    float sn = rs[(size_t)s * EDIM + e];
    float qrot = (e & 1) ? qsh[e - 1] : -qsh[e + 1];
    float krot = (e & 1) ? ksh[e - 1] : -ksh[e + 1];
    qn = qn * c + qrot * sn;
    kn = kn * c + krot * sn;
  }
  qn *= 0.08838834764831845f;  // 1/sqrt(128) folded into q

  size_t base = ((size_t)h * STOT + s) * EDIM + e;
  Qf[base] = f2bf(qn);
  Kf[base] = f2bf(kn);
  Vt[((size_t)h * EDIM + e) * STOT + s] = f2bf(vv);
}

// ---------------------------------------------------------------------------
// Flash attention: one wave (32 threads) per (head, 16-query tile).
// Key tiles of 32; scores via 8 WMMAs; online softmax; P re-laid out via a
// 16x32 LDS tile; P@V via 8 WMMAs. K/V operands stream from L2.
// ---------------------------------------------------------------------------
__global__ __launch_bounds__(32) void flash_attn(
    const unsigned short* __restrict__ Qf,   // [H,STOT,E]
    const unsigned short* __restrict__ Kf,   // [H,STOT,E]
    const unsigned short* __restrict__ Vt,   // [H,E,STOT]
    unsigned short* __restrict__ Attn) {     // [STOT, DMODEL]
  __shared__ unsigned short Pl[16 * 32];

  int lane = threadIdx.x;
  int qt   = blockIdx.x;                     // 0..143
  int h    = blockIdx.y;                     // 0..23
  int qb   = qt * 16;
  int half = lane >> 4;
  int lcol = lane & 15;

  v16bf qa[4];
  const unsigned short* Qb = Qf + ((size_t)h * STOT + qb) * EDIM;
#pragma unroll
  for (int c = 0; c < 4; ++c) qa[c] = load_frag(Qb + c * 32, EDIM, lane);

  v8f acc[8];
#pragma unroll
  for (int nt = 0; nt < 8; ++nt) acc[nt] = v8f_zero();
  float mrow[8], lrow[8];
#pragma unroll
  for (int r = 0; r < 8; ++r) { mrow[r] = -1e30f; lrow[r] = 0.f; }

  for (int kb = 0; kb < STOT; kb += 32) {
    v8f s0 = v8f_zero(), s1 = v8f_zero();
    const unsigned short* K0 = Kf + ((size_t)h * STOT + kb) * EDIM;
#pragma unroll
    for (int c = 0; c < 4; ++c) {
      v16bf b0 = load_frag(K0 + c * 32, EDIM, lane);
      v16bf b1 = load_frag(K0 + 16 * EDIM + c * 32, EDIM, lane);
      s0 = wmma_bf16(qa[c], b0, s0);
      s1 = wmma_bf16(qa[c], b1, s1);
    }

#pragma unroll
    for (int r = 0; r < 8; ++r) {
      float mx = fmaxf(s0[r], s1[r]);
#pragma unroll
      for (int m = 1; m < 16; m <<= 1) mx = fmaxf(mx, __shfl_xor(mx, m, 32));
      float mnew = fmaxf(mrow[r], mx);
      float scl  = __expf(mrow[r] - mnew);
      float p0   = __expf(s0[r] - mnew);
      float p1   = __expf(s1[r] - mnew);
      float rsum = p0 + p1;
#pragma unroll
      for (int m = 1; m < 16; m <<= 1) rsum += __shfl_xor(rsum, m, 32);
      lrow[r] = lrow[r] * scl + rsum;
      mrow[r] = mnew;
#pragma unroll
      for (int nt = 0; nt < 8; ++nt) acc[nt][r] *= scl;
      int prow = r + half * 8;
      Pl[prow * 32 + lcol]      = f2bf(p0);
      Pl[prow * 32 + 16 + lcol] = f2bf(p1);
    }
    __syncthreads();

    v16bf pa = load_frag(Pl, 32, lane);
#pragma unroll
    for (int nt = 0; nt < 8; ++nt) {
      const unsigned short* Vb = Vt + ((size_t)h * EDIM + nt * 16) * STOT + kb;
      v16bf bv = load_frag(Vb, STOT, lane);
      acc[nt] = wmma_bf16(pa, bv, acc[nt]);
    }
    __syncthreads();
  }

#pragma unroll
  for (int r = 0; r < 8; ++r) {
    float inv = 1.0f / lrow[r];
    int rowg  = qb + r + half * 8;
#pragma unroll
    for (int nt = 0; nt < 8; ++nt) {
      Attn[(size_t)rowg * DMODEL + h * EDIM + nt * 16 + lcol] =
          f2bf(acc[nt][r] * inv);
    }
  }
}

// ---------------------------------------------------------------------------
// Host launcher
// ---------------------------------------------------------------------------
extern "C" void kernel_launch(void* const* d_in, const int* in_sizes, int n_in,
                              void* d_out, int out_size, void* d_ws, size_t ws_size,
                              hipStream_t stream) {
  (void)in_sizes; (void)n_in; (void)out_size; (void)ws_size;

  const float* spatial = (const float*)d_in[0];
  const float* prompt  = (const float*)d_in[1];
  const float* w_qkv   = (const float*)d_in[2];
  const float* w_aqkv  = (const float*)d_in[3];
  const float* b_aqkv  = (const float*)d_in[4];
  const float* nqw     = (const float*)d_in[5];
  const float* nkw     = (const float*)d_in[6];
  const float* naqw    = (const float*)d_in[7];
  const float* nakw    = (const float*)d_in[8];
  const float* w_out   = (const float*)d_in[9];
  const float* b_out   = (const float*)d_in[10];
  const float* w_aout  = (const float*)d_in[11];
  const float* b_aout  = (const float*)d_in[12];
  const float* ropec   = (const float*)d_in[13];
  const float* ropes   = (const float*)d_in[14];
  // d_in[15] = trans_mat (rotation hardcoded), d_in[16] = N (hardcoded)

  char*  ws  = (char*)d_ws;
  size_t off = 0;
  auto take = [&](size_t bytes) -> char* {
    char* p = ws + off;
    off += (bytes + 255) & ~(size_t)255;
    return p;
  };

  unsigned short* sp_bf   = (unsigned short*)take((size_t)NSPA * DMODEL * 2);
  unsigned short* pr_bf   = (unsigned short*)take((size_t)LPRM * PDIM * 2);
  unsigned short* wqkv_t  = (unsigned short*)take((size_t)QK3 * DMODEL * 2);
  unsigned short* waqkv_t = (unsigned short*)take((size_t)QK3 * PDIM * 2);
  unsigned short* wout_t  = (unsigned short*)take((size_t)DMODEL * DMODEL * 2);
  unsigned short* waout_t = (unsigned short*)take((size_t)DMODEL * DMODEL * 2);
  float*          qkv_f   = (float*)take((size_t)NSPA * QK3 * 4);
  float*          aqkv_f  = (float*)take((size_t)LPRM * QK3 * 4);
  unsigned short* Qf      = (unsigned short*)take((size_t)H_HEADS * STOT * EDIM * 2);
  unsigned short* Kf      = (unsigned short*)take((size_t)H_HEADS * STOT * EDIM * 2);
  unsigned short* Vt      = (unsigned short*)take((size_t)H_HEADS * STOT * EDIM * 2);
  unsigned short* attn    = (unsigned short*)take((size_t)STOT * DMODEL * 2);

  float* out_sp = (float*)d_out;                               // [NSPA, DMODEL]
  float* out_pr = out_sp + (size_t)NSPA * DMODEL;              // [LPRM, DMODEL]

  // 1) cast activations to bf16
  {
    size_t n1 = (size_t)NSPA * DMODEL;
    cast_f32_bf16<<<dim3((unsigned)((n1 + 255) / 256)), dim3(256), 0, stream>>>(spatial, sp_bf, n1);
    size_t n2 = (size_t)LPRM * PDIM;
    cast_f32_bf16<<<dim3((unsigned)((n2 + 255) / 256)), dim3(256), 0, stream>>>(prompt, pr_bf, n2);
  }

  // 2) transpose-cast weights: W[K,N] -> Wt[N,K] bf16
  transpose_cast<<<dim3(QK3 / 32, DMODEL / 32), dim3(32, 8), 0, stream>>>(w_qkv,  wqkv_t,  DMODEL, QK3);
  transpose_cast<<<dim3(QK3 / 32, PDIM / 32),   dim3(32, 8), 0, stream>>>(w_aqkv, waqkv_t, PDIM,   QK3);
  transpose_cast<<<dim3(DMODEL / 32, DMODEL / 32), dim3(32, 8), 0, stream>>>(w_out,  wout_t,  DMODEL, DMODEL);
  transpose_cast<<<dim3(DMODEL / 32, DMODEL / 32), dim3(32, 8), 0, stream>>>(w_aout, waout_t, DMODEL, DMODEL);

  // 3) fused QKV GEMMs
  gemm_bf16<<<dim3(QK3 / BN, NSPA / BM), dim3(256), 0, stream>>>(
      sp_bf, wqkv_t, nullptr, qkv_f, NSPA, QK3, DMODEL, 0);
  gemm_bf16<<<dim3(QK3 / BN, LPRM / BM), dim3(256), 0, stream>>>(
      pr_bf, waqkv_t, b_aqkv, aqkv_f, LPRM, QK3, PDIM, 1);

  // 4) RMSNorm + RoPE epilogue -> Qf/Kf/Vt (bf16)
  qkv_epilogue<<<dim3(STOT, H_HEADS), dim3(128), 0, stream>>>(
      qkv_f, aqkv_f, nqw, nkw, naqw, nakw, ropec, ropes, Qf, Kf, Vt);

  // 5) joint flash attention -> attn (bf16, merged [STOT, DMODEL])
  flash_attn<<<dim3(STOT / 16, H_HEADS), dim3(32), 0, stream>>>(Qf, Kf, Vt, attn);

  // 6) output projections (fp32 results straight into d_out)
  gemm_bf16<<<dim3(DMODEL / BN, NSPA / BM), dim3(256), 0, stream>>>(
      attn, wout_t, b_out, out_sp, NSPA, DMODEL, DMODEL, 1);
  gemm_bf16<<<dim3(DMODEL / BN, LPRM / BM), dim3(256), 0, stream>>>(
      attn + (size_t)NSPA * DMODEL, waout_t, b_aout, out_pr, LPRM, DMODEL, DMODEL, 1);
}
